// FullAttentionResidual_1726576853594
// MI455X (gfx1250) — matrix operand order repaired and checked by reading, not compiled, not run on hardware
//
#include <hip/hip_runtime.h>

// ---------------------------------------------------------------------------
// MI455X (gfx1250) fused "attention over previous layer outputs".
// Shapes: n=12 prev layers, B*S = 4096 tokens, D=1024, H=16 heads, Dh=64.
// Key algebra: scores and values are linear in x, softmax weights sum to 1:
//   score_h(x) = (q_h^T Wk_h) . x + q_h.bk_h          -> tiny [.,1024]@[1024,16]
//   out_h      = Wv_h @ (sum_n a_n x_n) + bv_h        -> mix first, project once
// => ~20 GFLOP instead of ~206 GFLOP; memory-bound (~430 MiB HBM, ~19 us).
// All GEMMs use v_wmma_f32_16x16x32_bf16 (wave32, 16x16 tiles, K=32).
// ---------------------------------------------------------------------------

typedef __attribute__((ext_vector_type(16))) __bf16 v16bf;
typedef __attribute__((ext_vector_type(8)))  float  v8f;

union BF16x16 {
  v16bf v;
  uint4 q[2];
  __bf16 h[16];
};

__device__ __forceinline__ v8f wmma_bf16(const v16bf& a, const v16bf& b, const v8f& c) {
  // D = A(16x32 bf16) x B(32x16 bf16) + C(16x16 f32)
  return __builtin_amdgcn_wmma_f32_16x16x32_bf16(false, a, false, b, (short)0, c, false, false);
}

#define NPREV 12
#define NTOK  4096
#define DMODEL 1024
#define NHEAD 16
#define DHEAD 64

// workspace layout (bytes)
#define OFF_QK   0u            // bf16 [16][1024]            32768
#define OFF_QB   32768u        // f32  [16]                  64
#define OFF_WV   65536u        // bf16 [1024][1024]          2 MiB
#define OFF_WO   2162688u      // bf16 [1024][1024]          2 MiB
#define OFF_ATTN 4259840u      // f32  [12][4096][16]        3 MiB
#define OFF_VOUT 7405568u      // f32  [4096][1024]          16 MiB

// ---------------------------------------------------------------------------
// Kernel 1: fold pseudo-query of layer_idx into Wk:
//   qkT[h][d] = sum_e q[h][e] * Wk[h*64+e][d];  qb[h] = q[h].bk_h
// ---------------------------------------------------------------------------
__global__ void __launch_bounds__(256) k_qk(const int* layer_idx, const float* pq,
                                            const float* Wk, const float* bk,
                                            __bf16* qkT, float* qb) {
  int gid = blockIdx.x * blockDim.x + threadIdx.x;  // 16384 threads
  int h = gid >> 10, d = gid & 1023;
  const float* q = pq + (size_t)(*layer_idx) * (NHEAD * DHEAD) + h * DHEAD;
  float s = 0.f;
  for (int e = 0; e < DHEAD; ++e)
    s += q[e] * Wk[(size_t)(h * DHEAD + e) * DMODEL + d];
  qkT[h * DMODEL + d] = (__bf16)s;
  if (gid < NHEAD) {
    const float* qh = pq + (size_t)(*layer_idx) * (NHEAD * DHEAD) + gid * DHEAD;
    float sb = 0.f;
    for (int e = 0; e < DHEAD; ++e) sb += qh[e] * bk[gid * DHEAD + e];
    qb[gid] = sb;
  }
}

// ---------------------------------------------------------------------------
// Kernel 2: convert Wv, Wo to bf16 (row-major; rows are WMMA B-matrix columns)
// ---------------------------------------------------------------------------
__global__ void __launch_bounds__(256) k_cvt(const float* Wv, const float* Wo,
                                             __bf16* Wvb, __bf16* Wob) {
  int gid = blockIdx.x * blockDim.x + threadIdx.x;  // 262144 threads, 4 elems each
  float4 a = ((const float4*)Wv)[gid];
  float4 b = ((const float4*)Wo)[gid];
  union { __bf16 h[4]; uint2 u; } p;
  p.h[0] = (__bf16)a.x; p.h[1] = (__bf16)a.y; p.h[2] = (__bf16)a.z; p.h[3] = (__bf16)a.w;
  ((uint2*)Wvb)[gid] = p.u;
  p.h[0] = (__bf16)b.x; p.h[1] = (__bf16)b.y; p.h[2] = (__bf16)b.z; p.h[3] = (__bf16)b.w;
  ((uint2*)Wob)[gid] = p.u;
}

// ---------------------------------------------------------------------------
// Kernel 3: scores + softmax over the 12-layer axis -> attn[12][4096][16].
// One wave per 16-token tile; scores[16 tok][16 head] per layer accumulated
// with WMMA (A = x tile bf16, B = qkT). The f32 C layout puts all 12 layer
// scores for a (token,head) pair in ONE lane -> lane-local softmax.
// ---------------------------------------------------------------------------
__global__ void __launch_bounds__(256) k_scores(const float* prev, const __bf16* qkT,
                                                const float* qb, float* attn) {
  int wave = blockIdx.x * (blockDim.x >> 5) + (threadIdx.x >> 5);
  int lane = threadIdx.x & 31;
  int lo = lane & 15, hi = lane >> 4;
  int t0 = wave * 16;

  v8f z = {0.f, 0.f, 0.f, 0.f, 0.f, 0.f, 0.f, 0.f};
  v8f acc[NPREV];
#pragma unroll
  for (int n = 0; n < NPREV; ++n) acc[n] = z;

#pragma unroll
  for (int n = 0; n < NPREV; ++n) {
    const float* xr = prev + ((size_t)n * NTOK + t0 + lo) * DMODEL;
    for (int k0 = 0; k0 < DMODEL; k0 += 32) {
      // A: row M = lo, e0..7 = K k0+8*hi.., e8..15 = K k0+16+8*hi..
      const float4* p0 = (const float4*)(xr + k0 + 8 * hi);
      float4 a0 = p0[0], a1 = p0[1], a2 = p0[4], a3 = p0[5];
      BF16x16 A, B;
      A.h[0] = (__bf16)a0.x; A.h[1] = (__bf16)a0.y; A.h[2] = (__bf16)a0.z; A.h[3] = (__bf16)a0.w;
      A.h[4] = (__bf16)a1.x; A.h[5] = (__bf16)a1.y; A.h[6] = (__bf16)a1.z; A.h[7] = (__bf16)a1.w;
      A.h[8] = (__bf16)a2.x; A.h[9] = (__bf16)a2.y; A.h[10] = (__bf16)a2.z; A.h[11] = (__bf16)a2.w;
      A.h[12] = (__bf16)a3.x; A.h[13] = (__bf16)a3.y; A.h[14] = (__bf16)a3.z; A.h[15] = (__bf16)a3.w;
      // B: col N = lo (head), e0..7 = K k0+8*hi.., e8..15 = K k0+16+8*hi..
      const uint4* bq = (const uint4*)(qkT + lo * DMODEL + k0 + 8 * hi);
      B.q[0] = bq[0]; B.q[1] = bq[2];
      acc[n] = wmma_bf16(A.v, B.v, acc[n]);
    }
  }

  const float scale = 0.125f;  // Dh^-0.5, Dh = 64
  float qbh = qb[lo] * scale;
#pragma unroll
  for (int r = 0; r < 8; ++r) {
    float s[NPREV];
#pragma unroll
    for (int n = 0; n < NPREV; ++n) s[n] = acc[n][r] * scale + qbh;
    float m = s[0];
#pragma unroll
    for (int n = 1; n < NPREV; ++n) m = fmaxf(m, s[n]);
    float sum = 0.f;
#pragma unroll
    for (int n = 0; n < NPREV; ++n) { s[n] = __expf(s[n] - m); sum += s[n]; }
    float inv = 1.f / sum;
    int t = t0 + r + 8 * hi;  // C row = r + 8*(lane>>4)
#pragma unroll
    for (int n = 0; n < NPREV; ++n)
      attn[((size_t)n * NTOK + t) * NHEAD + lo] = s[n] * inv;
  }
}

// ---------------------------------------------------------------------------
// Kernel 4: per (16-token tile, head): xmix = sum_n a_n * x_n (f32 regs),
// stash as bf16 in LDS, then vout_h = xmix @ Wv_h^T + bv_h via WMMA.
// ---------------------------------------------------------------------------
__global__ void __launch_bounds__(256) k_mix(const float* prev, const float* attn,
                                             const __bf16* Wvb, const float* bv,
                                             float* vout) {
  __shared__ __bf16 xmix[16][1032];   // +16B row pad -> no LDS bank conflicts
  __shared__ float attnS[NPREV][16];
  int tile = blockIdx.x >> 4, gh = blockIdx.x & 15;
  int t0 = tile * 16;
  int tid = threadIdx.x;

  for (int i = tid; i < NPREV * 16; i += 256) {
    int n = i >> 4, t = i & 15;
    attnS[n][t] = attn[((size_t)n * NTOK + t0 + t) * NHEAD + gh];
  }
  __syncthreads();

  // Phase 1: register accumulation of the layer-mixed inputs (16 f32x4/thread)
  int t = tid >> 4, c = tid & 15;
  float4 acc[16];
#pragma unroll
  for (int k = 0; k < 16; ++k) acc[k] = make_float4(0.f, 0.f, 0.f, 0.f);
  for (int n = 0; n < NPREV; ++n) {
    float a = attnS[n][t];
    const float4* xr = (const float4*)(prev + ((size_t)n * NTOK + t0 + t) * DMODEL);
#pragma unroll
    for (int k = 0; k < 16; ++k) {
      float4 x = xr[c + 16 * k];
      acc[k].x += a * x.x; acc[k].y += a * x.y; acc[k].z += a * x.z; acc[k].w += a * x.w;
    }
  }
#pragma unroll
  for (int k = 0; k < 16; ++k) {
    union { __bf16 h[4]; uint2 u; } p;
    p.h[0] = (__bf16)acc[k].x; p.h[1] = (__bf16)acc[k].y;
    p.h[2] = (__bf16)acc[k].z; p.h[3] = (__bf16)acc[k].w;
    *(uint2*)&xmix[t][(c + 16 * k) * 4] = p.u;
  }
  __syncthreads();

  // Phase 2: [16x1024] @ [1024x64] per head; waves 0..3 take 16-col slices
  int w = tid >> 5;
  if (w < 4) {
    int lane = tid & 31, lo = lane & 15, hi = lane >> 4;
    int j = gh * DHEAD + w * 16 + lo;  // output column (value dim)
    v8f cacc = {0.f, 0.f, 0.f, 0.f, 0.f, 0.f, 0.f, 0.f};
    for (int k0 = 0; k0 < DMODEL; k0 += 32) {
      BF16x16 A, B;
      A.q[0] = *(const uint4*)&xmix[lo][k0 + 8 * hi];
      A.q[1] = *(const uint4*)&xmix[lo][k0 + 16 + 8 * hi];
      const uint4* bq = (const uint4*)(Wvb + (size_t)j * DMODEL + k0 + 8 * hi);
      B.q[0] = bq[0]; B.q[1] = bq[2];
      cacc = wmma_bf16(A.v, B.v, cacc);
    }
    float bvj = bv[j];
#pragma unroll
    for (int r = 0; r < 8; ++r)
      vout[(size_t)(t0 + r + 8 * hi) * DMODEL + j] = cacc[r] + bvj;
  }
}

// ---------------------------------------------------------------------------
// Kernel 5: out = vout @ Wo^T + bo.  LDS-staged A tile, 8 waves x 16 cols.
// ---------------------------------------------------------------------------
__global__ void __launch_bounds__(256) k_out(const float* vout, const __bf16* Wob,
                                             const float* bo, float* out) {
  __shared__ __bf16 As[16][1032];
  int tile = blockIdx.x >> 3, cg = blockIdx.x & 7;
  int t0 = tile * 16;
  int tid = threadIdx.x;

  int t = tid >> 4, c = tid & 15;
  const float4* vr = (const float4*)(vout + (size_t)(t0 + t) * DMODEL);
#pragma unroll
  for (int k = 0; k < 16; ++k) {
    float4 x = vr[c + 16 * k];
    union { __bf16 h[4]; uint2 u; } p;
    p.h[0] = (__bf16)x.x; p.h[1] = (__bf16)x.y; p.h[2] = (__bf16)x.z; p.h[3] = (__bf16)x.w;
    *(uint2*)&As[t][(c + 16 * k) * 4] = p.u;
  }
  __syncthreads();

  int w = tid >> 5, lane = tid & 31, lo = lane & 15, hi = lane >> 4;
  int o = cg * 128 + w * 16 + lo;  // output column
  v8f cacc = {0.f, 0.f, 0.f, 0.f, 0.f, 0.f, 0.f, 0.f};
  for (int k0 = 0; k0 < DMODEL; k0 += 32) {
    BF16x16 A, B;
    A.q[0] = *(const uint4*)&As[lo][k0 + 8 * hi];
    A.q[1] = *(const uint4*)&As[lo][k0 + 16 + 8 * hi];
    const uint4* bq = (const uint4*)(Wob + (size_t)o * DMODEL + k0 + 8 * hi);
    B.q[0] = bq[0]; B.q[1] = bq[2];
    cacc = wmma_bf16(A.v, B.v, cacc);
  }
  float boo = bo[o];
#pragma unroll
  for (int r = 0; r < 8; ++r)
    out[(size_t)(t0 + r + 8 * hi) * DMODEL + o] = cacc[r] + boo;
}

// ---------------------------------------------------------------------------
extern "C" void kernel_launch(void* const* d_in, const int* in_sizes, int n_in,
                              void* d_out, int out_size, void* d_ws, size_t ws_size,
                              hipStream_t stream) {
  (void)in_sizes; (void)n_in; (void)out_size; (void)ws_size;
  const int*   layer_idx = (const int*)d_in[0];
  const float* prev      = (const float*)d_in[1];
  /* d_in[2] current_output: unused by the reference */
  const float* pq        = (const float*)d_in[3];
  const float* Wk        = (const float*)d_in[4];
  const float* bk        = (const float*)d_in[5];
  const float* Wv        = (const float*)d_in[6];
  const float* bv        = (const float*)d_in[7];
  const float* Wo        = (const float*)d_in[8];
  const float* bo        = (const float*)d_in[9];

  char* ws = (char*)d_ws;
  __bf16* qkT  = (__bf16*)(ws + OFF_QK);
  float*  qb   = (float*) (ws + OFF_QB);
  __bf16* Wvb  = (__bf16*)(ws + OFF_WV);
  __bf16* Wob  = (__bf16*)(ws + OFF_WO);
  float*  attn = (float*) (ws + OFF_ATTN);
  float*  vout = (float*) (ws + OFF_VOUT);

  k_qk    <<<64,   256, 0, stream>>>(layer_idx, pq, Wk, bk, qkT, qb);
  k_cvt   <<<1024, 256, 0, stream>>>(Wv, Wo, Wvb, Wob);
  k_scores<<<32,   256, 0, stream>>>(prev, qkT, qb, attn);
  k_mix   <<<4096, 256, 0, stream>>>(prev, attn, Wvb, bv, vout);
  k_out   <<<2048, 256, 0, stream>>>(vout, Wob, bo, (float*)d_out);
}
